// Relation_4346506903992
// MI455X (gfx1250) — matrix operand order repaired and checked by reading, not compiled, not run on hardware
//
#include <hip/hip_runtime.h>
#include <cstdint>

#define DIMX   512
#define NQK    512
#define NV     512
#define NOUT   1536          // 512 q + 512 k + 512 v columns
#define HEADS  8
#define BB     4
#define NN     2048
#define ROWS   (BB*NN)       // 8192 tokens
#define DHEAD  64

typedef __attribute__((ext_vector_type(16))) _Float16 v16h;
typedef __attribute__((ext_vector_type(8)))  _Float16 v8h;
typedef __attribute__((ext_vector_type(8)))  float    v8f;
typedef __attribute__((ext_vector_type(2)))  float    v2f;
typedef __attribute__((ext_vector_type(4)))  unsigned int v4u;
typedef __attribute__((ext_vector_type(8)))  int      v8i_t;
typedef __attribute__((ext_vector_type(4)))  int      v4i_t;

static __device__ __forceinline__ v16h cat8(v8h lo, v8h hi) {
  return __builtin_shufflevector(lo, hi, 0,1,2,3,4,5,6,7,8,9,10,11,12,13,14,15);
}

// ---- TDM: load one 16-row x 64-col fp16 tile (row stride 512 elems) into LDS ----
// D# layout per CDNA5 ISA (group0 128b, group1 256b).
// 6-arg builtin form (clang-23 / amdgpu-toolchain).
static __device__ __forceinline__ void tdm_load_k_tile(const _Float16* gsrc,
                                                       unsigned int lds_byte_off) {
  const unsigned long long ga = (unsigned long long)(uintptr_t)gsrc;
  v4u g0;
  g0.x = 0x1u;                                        // count=1 (valid user D#)
  g0.y = lds_byte_off;                                // lds_addr  [63:32]
  g0.z = (unsigned int)(ga & 0xFFFFFFFFu);            // global_addr[95:64]
  g0.w = (unsigned int)((ga >> 32) & 0x01FFFFFFu)     // global_addr[120:96]
       | (2u << 30);                                  // type=2 ("image") [127:126]
  v8i_t g1;
  g1[0] = (int)(1u << 16);          // workgroup_mask=0, data_size=1 (2 bytes)
  g1[1] = (int)(512u << 16);        // tensor_dim0[15:0]=512 at bits[79:48]
  g1[2] = (int)(16u  << 16);        // tensor_dim0 hi=0 | tensor_dim1[15:0]=16
  g1[3] = (int)(64u  << 16);        // tensor_dim1 hi=0 | tile_dim0=64
  g1[4] = (int)(16u);               // tile_dim1=16, tile_dim2=0
  g1[5] = (int)(512u);              // tensor_dim0_stride[31:0]=512
  g1[6] = 0;                        // stride0 hi=0 | stride1 lo=0
  g1[7] = 0;                        // stride1 hi=0
  v4i_t g2 = {0, 0, 0, 0};
  v4i_t g3 = {0, 0, 0, 0};
  v8i_t g4 = {0, 0, 0, 0, 0, 0, 0, 0};
  __builtin_amdgcn_tensor_load_to_lds(g0, g1, g2, g3, g4, 0);
}

// ---------------- fp32 -> fp16 conversions ----------------
__global__ __launch_bounds__(256) void cvt_x_f16(const float* __restrict__ x,
                                                 _Float16* __restrict__ xh) {
  int i = blockIdx.x * 256 + threadIdx.x;           // ROWS*DIMX threads exactly
  xh[i] = (_Float16)x[i];
}

__global__ __launch_bounds__(256) void cvt_w_f16(const float* __restrict__ Wq,
                                                 const float* __restrict__ Wkv,
                                                 _Float16* __restrict__ Wt) {
  int i = blockIdx.x * 256 + threadIdx.x;           // NOUT*DIMX threads exactly
  int n = i >> 9;                                   // output col 0..1535
  int k = i & 511;                                  // input dim
  float w = (n < NQK) ? Wq[(size_t)k * NQK + n]
                      : Wkv[(size_t)k * (NQK + NV) + (n - NQK)];
  Wt[(size_t)n * DIMX + k] = (_Float16)w;
}

// ------- projection GEMM [8192x512] @ [512x1536]: 16x64 strip per wave -------
__global__ __launch_bounds__(256) void proj_wmma(const _Float16* __restrict__ xh,
                                                 const _Float16* __restrict__ Wt,
                                                 _Float16* __restrict__ qh,
                                                 _Float16* __restrict__ kh,
                                                 float* __restrict__ vf) {
  const int lane     = threadIdx.x & 31;
  const int wave     = threadIdx.x >> 5;
  const int tile     = blockIdx.x * 8 + wave;       // 0..12287 (exact grid)
  const int mt       = tile / (NOUT / 64);          // 0..511 row tile
  const int ntg      = tile % (NOUT / 64);          // 0..23  64-col group
  const int half_sel = lane >> 4;
  const int lm       = lane & 15;

  // A fragment (16x32 f16): low lanes K {0..15}, high lanes K {16..31}
  const _Float16* arow = xh + (size_t)(mt * 16 + lm) * DIMX + (half_sel ? 16 : 0);
  // B fragments (32x16 f16): lane holds a 32B contiguous K-run of one column
  const _Float16* brow0 = Wt + (size_t)(ntg * 64 +  0 + lm) * DIMX + (half_sel ? 16 : 0);
  const _Float16* brow1 = brow0 + (size_t)16 * DIMX;
  const _Float16* brow2 = brow0 + (size_t)32 * DIMX;
  const _Float16* brow3 = brow0 + (size_t)48 * DIMX;

  v8f acc0 = {}, acc1 = {}, acc2 = {}, acc3 = {};
#pragma unroll
  for (int kc = 0; kc < DIMX / 32; ++kc) {
    const int d = kc * 32;
    v16h a = *(const v16h*)(arow + d);
    v16h b0 = *(const v16h*)(brow0 + d);
    v16h b1 = *(const v16h*)(brow1 + d);
    v16h b2 = *(const v16h*)(brow2 + d);
    v16h b3 = *(const v16h*)(brow3 + d);
    acc0 = __builtin_amdgcn_wmma_f32_16x16x32_f16(false, a, false, b0, (short)0, acc0, false, false);
    acc1 = __builtin_amdgcn_wmma_f32_16x16x32_f16(false, a, false, b1, (short)0, acc1, false, false);
    acc2 = __builtin_amdgcn_wmma_f32_16x16x32_f16(false, a, false, b2, (short)0, acc2, false, false);
    acc3 = __builtin_amdgcn_wmma_f32_16x16x32_f16(false, a, false, b3, (short)0, acc3, false, false);
  }

  v8f accs[4] = {acc0, acc1, acc2, acc3};
#pragma unroll
  for (int g = 0; g < 4; ++g) {
    const int colBase = ntg * 64 + g * 16;           // wave-uniform
    const int ocol    = colBase + lm;
#pragma unroll
    for (int r = 0; r < 8; ++r) {
      const int   orow = mt * 16 + r + (half_sel ? 8 : 0);
      const float a    = accs[g][r];
      if (colBase < NQK) {
        qh[(size_t)orow * NQK + ocol] = (_Float16)(a * 0.125f);   // fold dq^-1/2
      } else if (colBase < 2 * NQK) {
        kh[(size_t)orow * NQK + (ocol - NQK)] = (_Float16)a;
      } else {
        vf[(size_t)orow * NV + (ocol - 2 * NQK)] = a;             // v stays fp32
      }
    }
  }
}

// ---- scores via WMMA; K tiles staged in LDS via TDM (double-buffered, shared by 8 waves) ----
__global__ __launch_bounds__(256) void attn_hard_wmma(const _Float16* __restrict__ qh,
                                                      const _Float16* __restrict__ kh,
                                                      const float* __restrict__ vf,
                                                      float* __restrict__ out) {
  __shared__ _Float16 kbuf[2][16 * DHEAD];           // 2 x 2KB double buffer
  __shared__ int rowIdx[8][16];
  const int lane     = threadIdx.x & 31;
  const int wave     = threadIdx.x >> 5;
  const int bh       = blockIdx.x >> 4;              // 0..31: all 8 waves share (b,h)
  const int qg       = blockIdx.x & 15;
  const int b        = bh >> 3;
  const int h        = bh & 7;
  const int qt       = qg * 8 + wave;                // wave-private q tile 0..127
  const int half_sel = lane >> 4;
  const int lm       = lane & 15;
  const int qBase    = qt * 16;

  // A fragments (q rows, d 0..31 / 32..63) loaded once, reused 128x.
  const _Float16* qp = qh + (size_t)(b * NN + qBase + lm) * NQK + h * DHEAD + (half_sel ? 16 : 0);
  v16h a0 = *(const v16h*)(qp);
  v16h a1 = *(const v16h*)(qp + 32);

  float best[8];
  int   bidx[8];
#pragma unroll
  for (int r = 0; r < 8; ++r) { best[r] = -3.0e38f; bidx[r] = 0; }

  const _Float16* kbase = kh + (size_t)(b * NN) * NQK + h * DHEAD;

  // Preload tile 0 via Tensor Data Mover.
  if (wave == 0) {
    tdm_load_k_tile(kbase, (unsigned int)(uintptr_t)&kbuf[0][0]);
    __builtin_amdgcn_s_wait_tensorcnt(0);
  }
  __syncthreads();

  for (int jt = 0; jt < NN / 16; ++jt) {
    const int cur = jt & 1;
    if (wave == 0 && jt + 1 < NN / 16) {
      tdm_load_k_tile(kbase + (size_t)(jt + 1) * 16 * NQK,
                      (unsigned int)(uintptr_t)&kbuf[cur ^ 1][0]);
    }
    // B fragments from LDS: lane = column (k row jBase+lm), 32B contiguous K-run
    const _Float16* kl = &kbuf[cur][lm * DHEAD + (half_sel ? 16 : 0)];
    v16h b0 = *(const v16h*)(kl);
    v16h b1 = *(const v16h*)(kl + 32);
    v8f c = {};
    c = __builtin_amdgcn_wmma_f32_16x16x32_f16(false, a0, false, b0, (short)0, c, false, false);
    c = __builtin_amdgcn_wmma_f32_16x16x32_f16(false, a1, false, b1, (short)0, c, false, false);
    const int j = jt * 16 + lm;
#pragma unroll
    for (int r = 0; r < 8; ++r) {
      const float s = c[r];
      if (s > best[r]) { best[r] = s; bidx[r] = j; }  // '>' keeps first occurrence
    }
    if (wave == 0 && jt + 1 < NN / 16) __builtin_amdgcn_s_wait_tensorcnt(0);
    __syncthreads();                                  // publish buf, close WAR on buf^1
  }

  // Cross-lane argmax within each 16-lane half (rows r and r+8 independently).
#pragma unroll
  for (int r = 0; r < 8; ++r) {
    float m  = best[r];
    int   id = bidx[r];
#pragma unroll
    for (int off = 8; off >= 1; off >>= 1) {
      float m2  = __shfl_xor(m, off, 32);
      int   id2 = __shfl_xor(id, off, 32);
      if (m2 > m || (m2 == m && id2 < id)) { m = m2; id = id2; }
    }
    if (lm == 0) rowIdx[wave][r + (half_sel ? 8 : 0)] = id;
  }
  __builtin_amdgcn_wave_barrier();   // same-wave LDS ordering; DS ops are in-order

  // Hard output: out[b, qBase+m, h*64+d] = v[b, j*, h*64+d] (fp32 exact)
#pragma unroll
  for (int m = 0; m < 16; ++m) {
    const int    js  = rowIdx[wave][m];
    const float* src = vf  + (size_t)(b * NN + js) * NV + h * DHEAD;
    float*       dst = out + (size_t)(b * NN + qBase + m) * (HEADS * DHEAD) + h * DHEAD;
    v2f t = *(const v2f*)(src + lane * 2);            // 32 lanes x 2 floats = 64
    *(v2f*)(dst + lane * 2) = t;
  }
}

extern "C" void kernel_launch(void* const* d_in, const int* in_sizes, int n_in,
                              void* d_out, int out_size, void* d_ws, size_t ws_size,
                              hipStream_t stream) {
  const float* x   = (const float*)d_in[0];   // [4, 2048, 512]
  const float* Wq  = (const float*)d_in[1];   // [512, 512]
  const float* Wkv = (const float*)d_in[2];   // [512, 1024]
  float* out = (float*)d_out;                 // [4, 2048, 512] fp32
  char*  ws  = (char*)d_ws;

  _Float16* xh = (_Float16*)(ws);                       //  8 MB
  _Float16* Wt = (_Float16*)(ws + ( 8u << 20));         //  1.5 MB
  _Float16* qh = (_Float16*)(ws + (10u << 20));         //  8 MB
  _Float16* kh = (_Float16*)(ws + (18u << 20));         //  8 MB
  float*    vf = (float*)   (ws + (26u << 20));         // 16 MB

  cvt_x_f16<<<(ROWS * DIMX) / 256, 256, 0, stream>>>(x, xh);
  cvt_w_f16<<<(NOUT * DIMX) / 256, 256, 0, stream>>>(Wq, Wkv, Wt);
  proj_wmma<<<((ROWS / 16) * (NOUT / 64)) / 8, 256, 0, stream>>>(xh, Wt, qh, kh, vf);
  attn_hard_wmma<<<((BB * HEADS) * (NN / 16 / 8)), 256, 0, stream>>>(qh, kh, vf, out);
}
